// GCNModelFeaAttScatStructureOnlyVAE_481036337863
// MI455X (gfx1250) — matrix operand order repaired and checked by reading, not compiled, not run on hardware
//
#include <hip/hip_runtime.h>

// ---------------------------------------------------------------------------
// Types for CDNA5 WMMA (gfx1250, wave32)
// ---------------------------------------------------------------------------
typedef __attribute__((ext_vector_type(16))) __bf16 v16bf;
typedef __attribute__((ext_vector_type(8)))  __bf16 v8bf;
typedef __attribute__((ext_vector_type(4)))  __bf16 v4bf;
typedef __attribute__((ext_vector_type(8)))  float  v8f;

union FragBF16 { v16bf v; v8bf h[2]; };

static constexpr int N_NODES = 8192;
static constexpr int D2 = 128;   // latent dim
static constexpr int D1 = 512;   // decoder width
static constexpr float BN_EPS = 1e-5f;

// ---------------------------------------------------------------------------
// Async global->LDS copy path (CDNA5 GLOBAL_LOAD_ASYNC_TO_LDS_B128, ASYNCcnt)
// Builtin signature (from clang diagnostic): arg0 = int4 addrspace(1)* (global
// source), arg1 = int4 addrspace(3)* (LDS dest), arg2 = imm offset, arg3 = cpol.
// ---------------------------------------------------------------------------
#if defined(__AMDGCN__) && __has_builtin(__builtin_amdgcn_global_load_async_to_lds_b128)
#define USE_ASYNC_G2L 1
typedef int v4i_vs __attribute__((vector_size(16)));
typedef __attribute__((address_space(1))) void   as1_void;
typedef __attribute__((address_space(3))) void   as3_void;
typedef __attribute__((address_space(1))) v4i_vs as1_v4i;
typedef __attribute__((address_space(3))) v4i_vs as3_v4i;
#else
#define USE_ASYNC_G2L 0
#endif

__device__ __forceinline__ void copy16_g2l(__bf16* l, const __bf16* g) {
#if USE_ASYNC_G2L
  __builtin_amdgcn_global_load_async_to_lds_b128(
      (as1_v4i*)(as1_void*)(void*)const_cast<__bf16*>(g),
      (as3_v4i*)(as3_void*)(void*)l, /*offset=*/0, /*cpol=*/0);
#else
  *(v8bf*)l = *(const v8bf*)g;
#endif
}

__device__ __forceinline__ void async_wait_all() {
#if USE_ASYNC_G2L
  asm volatile("s_wait_asynccnt 0x0" ::: "memory");
#endif
}

// ---------------------------------------------------------------------------
// Kernel 1: fused attention MLP + softmax gate + support = (x*score) @ W_gcn
// 16 rows per block so W matrices (3 x 256KB) are amortized over 16 rows.
// Writes support TRANSPOSED as bf16: supT[j][n], j in [0,512), n in [0,8192).
// ---------------------------------------------------------------------------
__global__ __launch_bounds__(256) void attn_support_kernel(
    const float* __restrict__ x, const float* __restrict__ W1,
    const float* __restrict__ W2, const float* __restrict__ Wg,
    __bf16* __restrict__ supT)
{
  constexpr int R = 16;
  const int n0 = blockIdx.x * R;
  const int t = threadIdx.x;

  __shared__ float xr[R][D2];
  __shared__ float hatt[R][D1];
  __shared__ float sc[R][D2];
  __shared__ float cond[R][D2];
  __shared__ float rowmax[R], rowinv[R];

  // load 16 rows of x
  for (int idx = t; idx < R * D2; idx += 256) {
    int r = idx >> 7, d = idx & 127;
    xr[r][d] = x[(size_t)(n0 + r) * D2 + d];
  }
  __syncthreads();

  // h_att = relu(x @ W1^T)  (W1: [512][128] row-major)
  for (int k = t; k < D1; k += 256) {
    const float* w = W1 + (size_t)k * D2;
    float acc[R];
#pragma unroll
    for (int r = 0; r < R; ++r) acc[r] = 0.f;
    for (int d = 0; d < D2; ++d) {
      float wd = w[d];
#pragma unroll
      for (int r = 0; r < R; ++r) acc[r] = fmaf(xr[r][d], wd, acc[r]);
    }
#pragma unroll
    for (int r = 0; r < R; ++r) hatt[r][k] = fmaxf(acc[r], 0.f);
  }
  __syncthreads();

  // scores = h_att @ W2^T  (W2: [128][512] row-major); one thread per j column
  if (t < D2) {
    const float* w = W2 + (size_t)t * D1;
    float acc[R];
#pragma unroll
    for (int r = 0; r < R; ++r) acc[r] = 0.f;
    for (int k = 0; k < D1; ++k) {
      float wd = w[k];
#pragma unroll
      for (int r = 0; r < R; ++r) acc[r] = fmaf(hatt[r][k], wd, acc[r]);
    }
#pragma unroll
    for (int r = 0; r < R; ++r) sc[r][t] = acc[r];
  }
  __syncthreads();

  // per-row softmax stats (rows are small: 128 elements)
  if (t < R) {
    float m = -1e30f;
    for (int j = 0; j < D2; ++j) m = fmaxf(m, sc[t][j]);
    float s = 0.f;
    for (int j = 0; j < D2; ++j) s += __expf(sc[t][j] - m);
    rowmax[t] = m;
    rowinv[t] = 1.f / s;
  }
  __syncthreads();

  // cond = x * softmax(score)
  for (int idx = t; idx < R * D2; idx += 256) {
    int r = idx >> 7, d = idx & 127;
    cond[r][d] = xr[r][d] * __expf(sc[r][d] - rowmax[r]) * rowinv[r];
  }
  __syncthreads();

  // support = cond @ Wg  (Wg: [128][512] row-major) -> write transposed bf16
  for (int j = t; j < D1; j += 256) {
    float acc[R];
#pragma unroll
    for (int r = 0; r < R; ++r) acc[r] = 0.f;
    for (int d = 0; d < D2; ++d) {
      float wd = Wg[(size_t)d * D1 + j];
#pragma unroll
      for (int r = 0; r < R; ++r) acc[r] = fmaf(cond[r][d], wd, acc[r]);
    }
#pragma unroll
    for (int r = 0; r < R; ++r)
      supT[(size_t)j * N_NODES + (n0 + r)] = (__bf16)acc[r];
  }
}

// ---------------------------------------------------------------------------
// WMMA GEMM: C[m][n] = sum_k A[m][k] * Bt[n][k]   (both K-contiguous)
// A is f32 (converted to bf16 while staging to LDS) or bf16; Bt is bf16.
// Block tile 128x128, 8 waves (4x2), wave tile 32x64, KT=64 stage.
// Double-buffered LDS; bf16 operands staged with GLOBAL_LOAD_ASYNC_TO_LDS
// (ASYNCcnt) so stage s+1 copies overlap stage s WMMAs.
// Fragment k order: contiguous k = 16*h + s for both A and B (wave32, 16x16x32 bf16).
// ---------------------------------------------------------------------------
template <bool RELU, typename AT>
__global__ __launch_bounds__(256) void wmma_gemm_kernel(
    const AT* __restrict__ A, const __bf16* __restrict__ Bt,
    float* __restrict__ C, int M, int Nn, int K, int ntilesN)
{
  constexpr int BM = 128, BN = 128, KT = 64, PAD = 8;
  __shared__ __bf16 As[2][BM][KT + PAD];
  __shared__ __bf16 Bs[2][BN][KT + PAD];

  const int tid = threadIdx.x;
  const int tn = blockIdx.x % ntilesN;   // N-tile inner -> consecutive blocks
  const int tm = blockIdx.x / ntilesN;   // share the same A panel in L2
  const int mbase = tm * BM, nbase = tn * BN;

  const int wv = tid >> 5, lane = tid & 31;
  const int wm = wv >> 1, wn = wv & 1;   // 4x2 wave grid
  const int l16 = lane & 15, lh = lane >> 4;

  v8f acc[2][4] = {};

  // stage global tiles for k-chunk k0 into LDS buffer `buf`
  auto stage_tiles = [&](int buf, int k0) {
    if constexpr (sizeof(AT) == 4) {
      // f32 source: load float4, convert to bf16 on the fly (sync path)
      for (int c = tid; c < BM * (KT / 4); c += 256) {
        int row = c >> 4;
        int kc = (c & 15) * 4;
        float4 v = *(const float4*)((const float*)A + (size_t)(mbase + row) * K + k0 + kc);
        v4bf o = { (__bf16)v.x, (__bf16)v.y, (__bf16)v.z, (__bf16)v.w };
        *(v4bf*)&As[buf][row][kc] = o;
      }
    } else {
      for (int c = tid; c < BM * (KT / 8); c += 256) {
        int row = c >> 3;
        int kc = (c & 7) * 8;
        copy16_g2l(&As[buf][row][kc],
                   (const __bf16*)A + (size_t)(mbase + row) * K + k0 + kc);
      }
    }
    for (int c = tid; c < BN * (KT / 8); c += 256) {
      int row = c >> 3;
      int kc = (c & 7) * 8;
      copy16_g2l(&Bs[buf][row][kc],
                 Bt + (size_t)(nbase + row) * K + k0 + kc);
    }
  };

  const int nstages = K / KT;
  stage_tiles(0, 0);

  for (int s = 0; s < nstages; ++s) {
    const int buf = s & 1;
    async_wait_all();     // this wave's async copies for stage s are done
    __syncthreads();      // all waves' copies (ds + async) visible

    if (s + 1 < nstages) stage_tiles(buf ^ 1, (s + 1) * KT);  // overlap

#pragma unroll
    for (int kk = 0; kk < KT; kk += 32) {
      FragBF16 a[2], b[4];
      // A frag: lane lh half: 16 contiguous K at lh*16
#pragma unroll
      for (int i = 0; i < 2; ++i) {
        const __bf16* p = &As[buf][wm * 32 + i * 16 + l16][kk + lh * 16];
        a[i].h[0] = *(const v8bf*)p;
        a[i].h[1] = *(const v8bf*)(p + 8);
      }
      // B frag: lane lh half: 16 contiguous K at lh*16 (Bt is [N][K]-major)
#pragma unroll
      for (int j = 0; j < 4; ++j) {
        const __bf16* p = &Bs[buf][wn * 64 + j * 16 + l16][kk + lh * 16];
        b[j].h[0] = *(const v8bf*)p;
        b[j].h[1] = *(const v8bf*)(p + 8);
      }
#pragma unroll
      for (int i = 0; i < 2; ++i)
#pragma unroll
        for (int j = 0; j < 4; ++j)
          acc[i][j] = __builtin_amdgcn_wmma_f32_16x16x32_bf16(
              false, a[i].v, false, b[j].v, (short)0, acc[i][j], false, false);
    }
    __syncthreads();  // all reads of As/Bs[buf] retired before it is re-filled
  }

  // epilogue: C v8f layout -> lane = n + 16*(m>>3), vgpr = m&7
#pragma unroll
  for (int i = 0; i < 2; ++i) {
#pragma unroll
    for (int j = 0; j < 4; ++j) {
      int mrow = mbase + wm * 32 + i * 16 + lh * 8;
      int ncol = nbase + wn * 64 + j * 16 + l16;
      float* cp = C + (size_t)mrow * Nn + ncol;
#pragma unroll
      for (int v = 0; v < 8; ++v) {
        float val = acc[i][j][v];
        if (RELU) val = fmaxf(val, 0.f);
        cp[(size_t)v * Nn] = val;
      }
    }
  }
}

// ---------------------------------------------------------------------------
// BatchNorm helpers
// ---------------------------------------------------------------------------
__global__ void zero_kernel(float* p, int n) {
  int i = blockIdx.x * blockDim.x + threadIdx.x;
  if (i < n) p[i] = 0.f;
}

__global__ __launch_bounds__(256) void bn_partial_kernel(
    const float* __restrict__ h, float* __restrict__ sums, float* __restrict__ sumsq)
{
  const int t = threadIdx.x;
  const int r0 = blockIdx.x * 128;   // 64 blocks x 128 rows
  float s0 = 0.f, s1 = 0.f, q0 = 0.f, q1 = 0.f;
  for (int r = r0; r < r0 + 128; ++r) {
    float a = h[(size_t)r * D1 + t];
    float b = h[(size_t)r * D1 + t + 256];
    s0 += a; q0 = fmaf(a, a, q0);
    s1 += b; q1 = fmaf(b, b, q1);
  }
  atomicAdd(&sums[t], s0);
  atomicAdd(&sums[t + 256], s1);
  atomicAdd(&sumsq[t], q0);
  atomicAdd(&sumsq[t + 256], q1);
}

__global__ void bn_finalize_kernel(
    const float* __restrict__ sums, const float* __restrict__ sumsq,
    const float* __restrict__ gamma, const float* __restrict__ beta,
    float* __restrict__ scale, float* __restrict__ shift)
{
  int j = threadIdx.x;  // 512 threads
  float mean = sums[j] * (1.f / (float)N_NODES);
  float var = sumsq[j] * (1.f / (float)N_NODES) - mean * mean;
  float rstd = rsqrtf(var + BN_EPS);
  float sc = gamma[j] * rstd;
  scale[j] = sc;
  shift[j] = beta[j] - mean * sc;
}

__global__ __launch_bounds__(256) void bn_apply_kernel(
    const float* __restrict__ h, const float* __restrict__ scale,
    const float* __restrict__ shift, __bf16* __restrict__ z)
{
  const size_t total4 = (size_t)N_NODES * D1 / 4;
  size_t stride = (size_t)gridDim.x * blockDim.x;
  for (size_t i = (size_t)blockIdx.x * blockDim.x + threadIdx.x; i < total4; i += stride) {
    float4 v = ((const float4*)h)[i];
    int j = (int)((i * 4) & (D1 - 1));
    v4bf o = { (__bf16)fmaf(v.x, scale[j + 0], shift[j + 0]),
               (__bf16)fmaf(v.y, scale[j + 1], shift[j + 1]),
               (__bf16)fmaf(v.z, scale[j + 2], shift[j + 2]),
               (__bf16)fmaf(v.w, scale[j + 3], shift[j + 3]) };
    ((v4bf*)z)[i] = o;
  }
}

// ---------------------------------------------------------------------------
// Launch
// ---------------------------------------------------------------------------
extern "C" void kernel_launch(void* const* d_in, const int* in_sizes, int n_in,
                              void* d_out, int out_size, void* d_ws, size_t ws_size,
                              hipStream_t stream) {
  (void)in_sizes; (void)n_in; (void)out_size; (void)ws_size;
  const float* x     = (const float*)d_in[0];
  const float* adj   = (const float*)d_in[1];
  const float* W1    = (const float*)d_in[2];
  const float* W2    = (const float*)d_in[3];
  const float* Wg    = (const float*)d_in[4];
  const float* gamma = (const float*)d_in[5];
  const float* beta  = (const float*)d_in[6];
  float* out = (float*)d_out;

  char* ws = (char*)d_ws;
  size_t off = 0;
  auto take = [&](size_t bytes) -> char* {
    char* p = ws + off;
    off += (bytes + 255) & ~(size_t)255;
    return p;
  };
  __bf16* supT  = (__bf16*)take((size_t)D1 * N_NODES * 2);       // 8 MB  [512][8192]
  float*  h     = (float*) take((size_t)N_NODES * D1 * 4);       // 16 MB [8192][512]
  __bf16* z     = (__bf16*)take((size_t)N_NODES * D1 * 2);       // 8 MB  [8192][512]
  float*  stats = (float*) take((size_t)4 * D1 * 4);             // sums|sumsq|scale|shift
  float* sums  = stats;
  float* sumsq = stats + D1;
  float* scale = stats + 2 * D1;
  float* shift = stats + 3 * D1;

  // 1) attention gate + support^T (bf16)
  attn_support_kernel<<<N_NODES / 16, 256, 0, stream>>>(x, W1, W2, Wg, supT);

  // 2) h = relu(adj @ support): M=8192, Nn=512, K=8192; adj converted in-tile
  wmma_gemm_kernel<true, float><<<(N_NODES / 128) * (D1 / 128), 256, 0, stream>>>(
      adj, supT, h, N_NODES, D1, N_NODES, D1 / 128);

  // 3) BatchNorm (training-mode batch stats, biased variance)
  zero_kernel<<<4, 256, 0, stream>>>(sums, 2 * D1);
  bn_partial_kernel<<<N_NODES / 128, 256, 0, stream>>>(h, sums, sumsq);
  bn_finalize_kernel<<<1, D1, 0, stream>>>(sums, sumsq, gamma, beta, scale, shift);
  bn_apply_kernel<<<2048, 256, 0, stream>>>(h, scale, shift, z);

  // 4) out = z @ z^T: M=Nn=8192, K=512; A and Bt are the same z buffer
  wmma_gemm_kernel<false, __bf16><<<(N_NODES / 128) * (N_NODES / 128), 256, 0, stream>>>(
      z, z, out, N_NODES, N_NODES, D1, N_NODES / 128);
}
